// S2T_G_64098091925731
// MI455X (gfx1250) — matrix operand order repaired and checked by reading, not compiled, not run on hardware
//
#include <hip/hip_runtime.h>
#include <hip/hip_bf16.h>
#include <math.h>

// ---------------------------------------------------------------------------
// Problem constants (fixed by the reference)
// ---------------------------------------------------------------------------
#define BB   128
#define VV   3587
#define TT   96
#define EMB  300
#define HID  256
#define TEMP 0.5f
#define EPS  1e-20f

// K paddings (multiples of 32 for WMMA 16x16x32)
#define KGATES 576    // 300 + 256 -> 576
#define KFEED  3616   // 3587 -> 3616 (113*32)

typedef __bf16 v16bf __attribute__((ext_vector_type(16)));
typedef float  v8f   __attribute__((ext_vector_type(8)));
typedef unsigned int u32x4 __attribute__((ext_vector_type(4)));
typedef int          i32x8 __attribute__((ext_vector_type(8)));
typedef int          i32x4 __attribute__((ext_vector_type(4)));

#if defined(__has_builtin)
#if __has_builtin(__builtin_amdgcn_tensor_load_to_lds) && __has_builtin(__builtin_amdgcn_s_wait_tensorcnt)
#define HAS_TDM 1
#endif
#endif

__device__ __forceinline__ float sigmoidf_(float x) { return 1.0f / (1.0f + __expf(-x)); }

#define V8F_ZERO {0.f, 0.f, 0.f, 0.f, 0.f, 0.f, 0.f, 0.f}

// ---------------------------------------------------------------------------
// Elementwise / packing kernels
// ---------------------------------------------------------------------------
__global__ void k_f32_to_bf16(const float* __restrict__ src,
                              __hip_bfloat16* __restrict__ dst, long n) {
  long i = (long)blockIdx.x * blockDim.x + threadIdx.x;
  if (i < n) dst[i] = __float2bfloat16(src[i]);
}

__global__ void k_pack_gates(const float* __restrict__ w_ih,
                             const float* __restrict__ w_hh,
                             __hip_bfloat16* __restrict__ wg) {
  long i = (long)blockIdx.x * blockDim.x + threadIdx.x;
  if (i >= (long)KGATES * 1024) return;
  int k = (int)(i / 1024), n = (int)(i % 1024);
  float v = 0.0f;
  if (k < 300)      v = w_ih[(long)n * 300 + k];
  else if (k < 556) v = w_hh[(long)n * 256 + (k - 300)];
  wg[i] = __float2bfloat16(v);
}

__global__ void k_bias_sum(const float* __restrict__ a, const float* __restrict__ b,
                           float* __restrict__ o, int n) {
  int i = blockIdx.x * blockDim.x + threadIdx.x;
  if (i < n) o[i] = a[i] + b[i];
}

__global__ void k_pack_wlin(const float* __restrict__ w_lin,
                            __hip_bfloat16* __restrict__ wt) {
  long i = (long)blockIdx.x * blockDim.x + threadIdx.x;
  if (i >= (long)HID * VV) return;
  int k = (int)(i / VV), n = (int)(i % VV);
  wt[i] = __float2bfloat16(w_lin[(long)n * HID + k]);
}

__global__ void k_pack_vocab(const float* __restrict__ emb,
                             __hip_bfloat16* __restrict__ vb) {
  long i = (long)blockIdx.x * blockDim.x + threadIdx.x;
  if (i >= (long)KFEED * EMB) return;
  int v = (int)(i / EMB), e = (int)(i % EMB);
  float x = (v < VV) ? emb[(long)v * EMB + e] : 0.0f;
  vb[i] = __float2bfloat16(x);
}

__global__ void k_init_state(const float* __restrict__ bos,
                             const float* __restrict__ latent,
                             __hip_bfloat16* __restrict__ inpcat,
                             float* __restrict__ cbuf) {
  int idx = blockIdx.x * blockDim.x + threadIdx.x;
  if (idx >= BB * KGATES) return;
  int b = idx / KGATES, j = idx % KGATES;
  float v = 0.0f;
  if (j < 300) v = bos[j];
  else if (j < 556) { v = latent[b * HID + (j - 300)]; cbuf[b * HID + (j - 300)] = v; }
  inpcat[idx] = __float2bfloat16(v);
}

// ---------------------------------------------------------------------------
// Implicit-GEMM Conv3D: one wave per 64(outch) x 16(position) macro-tile.
// One gathered B fragment is reused by 4 WMMAs (4 out-channel sub-tiles).
// ---------------------------------------------------------------------------
__global__ void k_conv3d_wmma(const __hip_bfloat16* __restrict__ srcH,
                              const __hip_bfloat16* __restrict__ wtH,
                              const float* __restrict__ bias,
                              __hip_bfloat16* __restrict__ dstH,
                              float* __restrict__ dstF,
                              int Cin, int Din, int Hin, int Win,
                              int Cout, int Dout, int Hout, int Wout,
                              int stride, int leaky) {
  const __bf16* src = reinterpret_cast<const __bf16*>(srcH);
  const __bf16* wt  = reinterpret_cast<const __bf16*>(wtH);
  const int lane = threadIdx.x;           // 0..31, wave32
  const int K    = Cin * 64;
  const int HWo  = Hout * Wout;
  const int Pout = Dout * HWo;
  const int HWin = Hin * Win;
  const int PinC = Din * HWin;

  const long n = (long)blockIdx.x * 16 + (lane & 15);   // output position id
  const int  m0 = blockIdx.y * 64;                       // 4 x 16 out-channel tiles
  const int  b  = (int)(n / Pout);
  const int  p  = (int)(n % Pout);
  const int  od = p / HWo;
  const int  rp = p % HWo;
  const int  oh = rp / Wout;
  const int  ow = rp % Wout;
  const long binbase = (long)b * Cin * PinC +
                       (long)(od * stride) * HWin + (oh * stride) * Win + (ow * stride);

  const __bf16* a0 = wt + (long)(m0 + (lane & 15)) * K;
  const int koffA = (lane >> 4) * 8;      // A: lanes 16-31 hold K+8 / K+24 halves
  const int koffB = (lane >> 4) * 8;      // B: same as A, lanes 16-31 hold K+8 / K+24 halves

  v8f acc[4] = {V8F_ZERO, V8F_ZERO, V8F_ZERO, V8F_ZERO};
  for (int k0 = 0; k0 < K; k0 += 32) {
    if (k0 + 32 < K) __builtin_prefetch(&a0[k0 + 32], 0, 1);
    v16bf bf;
#pragma unroll
    for (int i = 0; i < 16; ++i) {
      const int k  = k0 + koffB + i + (i & 8);
      const int ic = k >> 6;
      const int r  = k & 63;
      bf[i] = src[binbase + (long)ic * PinC + (r >> 4) * HWin + ((r >> 2) & 3) * Win + (r & 3)];
    }
#pragma unroll
    for (int s = 0; s < 4; ++s) {
      const __bf16* ar = a0 + (long)s * 16 * K;
      v16bf a;
#pragma unroll
      for (int j = 0; j < 8; ++j) {
        a[j]     = ar[k0 + koffA + j];
        a[j + 8] = ar[k0 + 16 + koffA + j];
      }
      acc[s] = __builtin_amdgcn_wmma_f32_16x16x32_bf16(false, a, false, bf, (short)0,
                                                       acc[s], false, false);
    }
  }

  const int mrow = (lane >> 4) * 8;       // C layout: VGPR r -> M = r (+8 for hi lanes)
#pragma unroll
  for (int s = 0; s < 4; ++s) {
#pragma unroll
    for (int r = 0; r < 8; ++r) {
      const int ch = m0 + s * 16 + mrow + r;
      float v = acc[s][r] + bias[ch];
      if (leaky) v = (v > 0.0f) ? v : 0.2f * v;
      const long oidx = ((long)b * Cout + ch) * Pout + p;
      if (dstH) dstH[oidx] = __float2bfloat16(v);
      if (dstF) dstF[oidx] = v;
    }
  }
}

// ---------------------------------------------------------------------------
// Decoder GEMM with TDM-staged B panel:
//   grid.x = N/16 tiles; block = 256 threads = 8 waves = full M=128.
//   Wave 0 issues tensor_load_to_lds for the [K x 16] B panel (bf16), waits
//   on TENSORcnt, then all 8 waves consume it from LDS.
// ---------------------------------------------------------------------------
__global__ void k_gemm_wmma_tdm(const __hip_bfloat16* __restrict__ Ah, int lda,
                                const __hip_bfloat16* __restrict__ Bh, int ldb,
                                int K, int N,
                                const float* __restrict__ bias,
                                float* __restrict__ outF,
                                __hip_bfloat16* __restrict__ outB, int ldc) {
  extern __shared__ __hip_bfloat16 smem[];          // K*16 bf16
  const int tid  = threadIdx.x;
  const int lane = tid & 31;
  const int wv   = tid >> 5;                        // 0..7 -> M tile
  const int n0   = blockIdx.x * 16;

#ifdef HAS_TDM
  if (wv == 0) {
    const unsigned long ga = (unsigned long)(uintptr_t)(Bh + n0);   // tile start (bytes)
    const unsigned lds_off = (unsigned)(uintptr_t)smem;
    // D# group0: count=1, lds_addr, 57-bit global addr, type=2 ("image")
    u32x4 g0 = { 1u,
                 lds_off,
                 (unsigned)(ga & 0xFFFFFFFFu),
                 (unsigned)((ga >> 32) & 0x1FFFFFFu) | (2u << 30) };
    // D# group1: wg_mask=0, data_size=1 (2B); tensor_dim0=N, tensor_dim1=K;
    //            tile_dim0=16, tile_dim1=K; tensor_dim0_stride=ldb
    i32x8 g1 = { (int)(1u << 16),
                 (int)(((unsigned)N & 0xFFFFu) << 16),
                 (int)((((unsigned)N >> 16) & 0xFFFFu) | (((unsigned)K & 0xFFFFu) << 16)),
                 (int)((((unsigned)K >> 16) & 0xFFFFu) | (16u << 16)),
                 (int)((unsigned)K & 0xFFFFu),
                 (int)ldb,
                 0, 0 };
    i32x4 gz4 = {0, 0, 0, 0};
    i32x8 gz8 = {0, 0, 0, 0, 0, 0, 0, 0};
    __builtin_amdgcn_tensor_load_to_lds(g0, g1, gz4, gz4, gz8, 0);
    __builtin_amdgcn_s_wait_tensorcnt(0);
  }
#else
  for (int i = tid; i < K * 16; i += 256) {
    const int kk = i >> 4, nn = i & 15;
    const int n = n0 + nn;
    smem[i] = (n < N) ? Bh[(long)kk * ldb + n] : __float2bfloat16(0.0f);
  }
#endif
  __syncthreads();

  const __bf16* A  = reinterpret_cast<const __bf16*>(Ah);
  const __bf16* lb = reinterpret_cast<const __bf16*>(smem);
  const int nl = lane & 15;
  const int n  = n0 + nl;
  const bool nok = (n < N);
  const int m0 = wv * 16;
  const __bf16* arow = A + (long)(m0 + nl) * lda;
  const int koffA = (lane >> 4) * 8;
  const int koffB = (lane >> 4) * 8;

  v8f acc = V8F_ZERO;
  for (int k0 = 0; k0 < K; k0 += 32) {
    v16bf a, bf;
#pragma unroll
    for (int j = 0; j < 8; ++j) {
      a[j]     = arow[k0 + koffA + j];
      a[j + 8] = arow[k0 + 16 + koffA + j];
    }
#pragma unroll
    for (int i = 0; i < 16; ++i) {
      bf[i] = lb[(k0 + koffB + i + (i & 8)) * 16 + nl];       // DS reads of staged panel
    }
    acc = __builtin_amdgcn_wmma_f32_16x16x32_bf16(false, a, false, bf, (short)0, acc,
                                                  false, false);
  }

  const float bv = (bias && nok) ? bias[n] : 0.0f;
  const int mrow = (lane >> 4) * 8;
#pragma unroll
  for (int r = 0; r < 8; ++r) {
    const int m = m0 + mrow + r;
    const float v = acc[r] + bv;
    if (nok) {
      if (outF) outF[(long)m * ldc + n] = v;
      if (outB) outB[(long)m * ldc + n] = __float2bfloat16(v);
    }
  }
}

// ---------------------------------------------------------------------------
// Plain global-operand WMMA GEMM (used for the feedback GEMM, K=3616).
// ---------------------------------------------------------------------------
__global__ void k_gemm_wmma(const __hip_bfloat16* __restrict__ Ah, int lda,
                            const __hip_bfloat16* __restrict__ Bh, int ldb,
                            int K, int N,
                            const float* __restrict__ bias,
                            float* __restrict__ outF,
                            __hip_bfloat16* __restrict__ outB, int ldc) {
  const __bf16* A = reinterpret_cast<const __bf16*>(Ah);
  const __bf16* B = reinterpret_cast<const __bf16*>(Bh);
  const int lane = threadIdx.x;
  const int n0 = blockIdx.x * 16;
  const int m0 = blockIdx.y * 16;
  const int nl = lane & 15;
  const int n  = n0 + nl;
  const bool nok = (n < N);

  const __bf16* arow = A + (long)(m0 + nl) * lda;
  const int koffA = (lane >> 4) * 8;
  const int koffB = (lane >> 4) * 8;

  v8f acc = V8F_ZERO;
  for (int k0 = 0; k0 < K; k0 += 32) {
    if (k0 + 32 < K) __builtin_prefetch(&B[(long)(k0 + 32 + koffB) * ldb + n], 0, 1);
    v16bf a, bf;
#pragma unroll
    for (int j = 0; j < 8; ++j) {
      a[j]     = arow[k0 + koffA + j];
      a[j + 8] = arow[k0 + 16 + koffA + j];
    }
#pragma unroll
    for (int i = 0; i < 16; ++i) {
      bf[i] = nok ? B[(long)(k0 + koffB + i + (i & 8)) * ldb + n] : (__bf16)0.0f;
    }
    acc = __builtin_amdgcn_wmma_f32_16x16x32_bf16(false, a, false, bf, (short)0, acc,
                                                  false, false);
  }

  const float bv = (bias && nok) ? bias[n] : 0.0f;
  const int mrow = (lane >> 4) * 8;
#pragma unroll
  for (int r = 0; r < 8; ++r) {
    const int m = m0 + mrow + r;
    const float v = acc[r] + bv;
    if (nok) {
      if (outF) outF[(long)m * ldc + n] = v;
      if (outB) outB[(long)m * ldc + n] = __float2bfloat16(v);
    }
  }
}

// ---------------------------------------------------------------------------
// LSTM cell elementwise
// ---------------------------------------------------------------------------
__global__ void k_lstm_cell(const float* __restrict__ gates,
                            float* __restrict__ cbuf,
                            __hip_bfloat16* __restrict__ inpcat) {
  int idx = blockIdx.x * blockDim.x + threadIdx.x;   // BB*HID
  if (idx >= BB * HID) return;
  int b = idx >> 8, j = idx & 255;
  const float* g = gates + (long)b * 1024;
  float ig = sigmoidf_(g[j]);
  float fg = sigmoidf_(g[256 + j]);
  float gg = tanhf(g[512 + j]);
  float og = sigmoidf_(g[768 + j]);
  float c  = fg * cbuf[idx] + ig * gg;
  float h  = og * tanhf(c);
  cbuf[idx] = c;
  inpcat[(long)b * KGATES + 300 + j] = __float2bfloat16(h);
}

// ---------------------------------------------------------------------------
// Gumbel-softmax over V per batch row (in-place z over logits).
// ---------------------------------------------------------------------------
__global__ void k_gumbel_softmax(float* __restrict__ logits,         // [BB, VV]
                                 const float* __restrict__ gumbel_u, // [BB, TT, VV]
                                 int t,
                                 float* __restrict__ out,            // [BB, TT, VV]
                                 __hip_bfloat16* __restrict__ gbf) { // [BB, KFEED]
  __shared__ float sred[256];
  __shared__ int   sidx[256];
  const int b = blockIdx.x, tid = threadIdx.x;
  float* lg = logits + (long)b * VV;
  const float* gu = gumbel_u + ((long)b * TT + t) * VV;

  float m1 = -3.4e38f;
  for (int v = tid; v < VV; v += 256) m1 = fmaxf(m1, lg[v]);
  sred[tid] = m1; __syncthreads();
  for (int o = 128; o > 0; o >>= 1) {
    if (tid < o) sred[tid] = fmaxf(sred[tid], sred[tid + o]);
    __syncthreads();
  }
  m1 = sred[0]; __syncthreads();

  float s1 = 0.0f;
  for (int v = tid; v < VV; v += 256) s1 += __expf(lg[v] - m1);
  sred[tid] = s1; __syncthreads();
  for (int o = 128; o > 0; o >>= 1) {
    if (tid < o) sred[tid] += sred[tid + o];
    __syncthreads();
  }
  const float lse = m1 + __logf(sred[0]); __syncthreads();

  float m2 = -3.4e38f; int am = 0;
  for (int v = tid; v < VV; v += 256) {
    float g = -__logf(-__logf(gu[v] + EPS) + EPS);
    float z = ((lg[v] - lse) + g) * (1.0f / TEMP);
    lg[v] = z;
    if (z > m2) { m2 = z; am = v; }
  }
  sred[tid] = m2; sidx[tid] = am; __syncthreads();
  for (int o = 128; o > 0; o >>= 1) {
    if (tid < o && sred[tid + o] > sred[tid]) { sred[tid] = sred[tid + o]; sidx[tid] = sidx[tid + o]; }
    __syncthreads();
  }
  m2 = sred[0]; am = sidx[0]; __syncthreads();

  float s2 = 0.0f;
  for (int v = tid; v < VV; v += 256) s2 += __expf(lg[v] - m2);
  sred[tid] = s2; __syncthreads();
  for (int o = 128; o > 0; o >>= 1) {
    if (tid < o) sred[tid] += sred[tid + o];
    __syncthreads();
  }
  const float inv = 1.0f / sred[0]; __syncthreads();

  float* orow = out + ((long)b * TT + t) * VV;
  __hip_bfloat16* grow = gbf + (long)b * KFEED;
  for (int v = tid; v < VV; v += 256) {
    float y = __expf(lg[v] - m2) * inv;
    float hard = (v == am) ? 1.0f : 0.0f;
    float gout = (hard - y) + y;
    orow[v] = gout;
    grow[v] = __float2bfloat16(gout);
  }
  for (int v = VV + tid; v < KFEED; v += 256) grow[v] = __float2bfloat16(0.0f);
}

// ---------------------------------------------------------------------------
// Host launcher
// ---------------------------------------------------------------------------
extern "C" void kernel_launch(void* const* d_in, const int* in_sizes, int n_in,
                              void* d_out, int out_size, void* d_ws, size_t ws_size,
                              hipStream_t stream) {
  (void)in_sizes; (void)n_in; (void)out_size; (void)ws_size;
  const float* x    = (const float*)d_in[0];
  const float* vemb = (const float*)d_in[1];
  const float* bos  = (const float*)d_in[2];
  const float* wc1  = (const float*)d_in[3];
  const float* bc1  = (const float*)d_in[4];
  const float* wc2  = (const float*)d_in[5];
  const float* bc2  = (const float*)d_in[6];
  const float* wc3  = (const float*)d_in[7];
  const float* bc3  = (const float*)d_in[8];
  const float* wc4  = (const float*)d_in[9];
  const float* bc4  = (const float*)d_in[10];
  const float* wih  = (const float*)d_in[11];
  const float* whh  = (const float*)d_in[12];
  const float* bih  = (const float*)d_in[13];
  const float* bhh  = (const float*)d_in[14];
  const float* wlin = (const float*)d_in[15];
  const float* blin = (const float*)d_in[16];
  const float* gum  = (const float*)d_in[17];
  float* out = (float*)d_out;

  char* wsb = (char*)d_ws;
  size_t off = 0;
  auto alloc = [&](size_t bytes) -> char* {
    off = (off + 255) & ~(size_t)255;
    char* p = wsb + off; off += bytes; return p;
  };
  const long NX  = (long)BB * 4 * 32 * 32 * 32;
  const long NA1 = (long)BB * 64  * 29 * 29 * 29;
  const long NA2 = (long)BB * 128 * 13 * 13 * 13;
  const long NA3 = (long)BB * 256 * 5 * 5 * 5;

  __hip_bfloat16* xbf   = (__hip_bfloat16*)alloc(NX * 2);
  __hip_bfloat16* c1bf  = (__hip_bfloat16*)alloc((long)64 * 256 * 2);
  __hip_bfloat16* c2bf  = (__hip_bfloat16*)alloc((long)128 * 4096 * 2);
  __hip_bfloat16* c3bf  = (__hip_bfloat16*)alloc((long)256 * 8192 * 2);
  __hip_bfloat16* c4bf  = (__hip_bfloat16*)alloc((long)256 * 16384 * 2);
  __hip_bfloat16* a1    = (__hip_bfloat16*)alloc(NA1 * 2);
  __hip_bfloat16* a2    = (__hip_bfloat16*)alloc(NA2 * 2);
  __hip_bfloat16* a3    = (__hip_bfloat16*)alloc(NA3 * 2);
  float*          latent= (float*)alloc((long)BB * HID * 4);
  __hip_bfloat16* wg    = (__hip_bfloat16*)alloc((long)KGATES * 1024 * 2);
  float*          biasg = (float*)alloc(1024 * 4);
  __hip_bfloat16* wlt   = (__hip_bfloat16*)alloc((long)HID * VV * 2);
  __hip_bfloat16* vbf   = (__hip_bfloat16*)alloc((long)KFEED * EMB * 2);
  __hip_bfloat16* inpcat= (__hip_bfloat16*)alloc((long)BB * KGATES * 2);
  float*          cbuf  = (float*)alloc((long)BB * HID * 4);
  float*          gates = (float*)alloc((long)BB * 1024 * 4);
  float*          logit = (float*)alloc((long)BB * VV * 4);
  __hip_bfloat16* gbf   = (__hip_bfloat16*)alloc((long)BB * KFEED * 2);

  auto gblk = [](long n) { return (unsigned)((n + 255) / 256); };

  // --- pack everything to bf16 ---
  k_f32_to_bf16<<<gblk(NX), 256, 0, stream>>>(x, xbf, NX);
  k_f32_to_bf16<<<gblk(64L * 256), 256, 0, stream>>>(wc1, c1bf, 64L * 256);
  k_f32_to_bf16<<<gblk(128L * 4096), 256, 0, stream>>>(wc2, c2bf, 128L * 4096);
  k_f32_to_bf16<<<gblk(256L * 8192), 256, 0, stream>>>(wc3, c3bf, 256L * 8192);
  k_f32_to_bf16<<<gblk(256L * 16384), 256, 0, stream>>>(wc4, c4bf, 256L * 16384);
  k_pack_gates<<<gblk((long)KGATES * 1024), 256, 0, stream>>>(wih, whh, wg);
  k_bias_sum<<<4, 256, 0, stream>>>(bih, bhh, biasg, 1024);
  k_pack_wlin<<<gblk((long)HID * VV), 256, 0, stream>>>(wlin, wlt);
  k_pack_vocab<<<gblk((long)KFEED * EMB), 256, 0, stream>>>(vemb, vbf);

  // --- conv encoder (implicit GEMM, 4x M-blocked WMMA) ---
  k_conv3d_wmma<<<dim3((unsigned)(BB * 24389L / 16), 1), 32, 0, stream>>>(
      xbf, c1bf, bc1, a1, nullptr, 4, 32, 32, 32, 64, 29, 29, 29, 1, 1);
  k_conv3d_wmma<<<dim3((unsigned)(BB * 2197L / 16), 2), 32, 0, stream>>>(
      a1, c2bf, bc2, a2, nullptr, 64, 29, 29, 29, 128, 13, 13, 13, 2, 1);
  k_conv3d_wmma<<<dim3((unsigned)(BB * 125L / 16), 4), 32, 0, stream>>>(
      a2, c3bf, bc3, a3, nullptr, 128, 13, 13, 13, 256, 5, 5, 5, 2, 0);
  k_conv3d_wmma<<<dim3((unsigned)(BB / 16), 4), 32, 0, stream>>>(
      a3, c4bf, bc4, nullptr, latent, 256, 5, 5, 5, 256, 1, 1, 1, 2, 0);

  // --- decoder init ---
  k_init_state<<<gblk((long)BB * KGATES), 256, 0, stream>>>(bos, latent, inpcat, cbuf);

  // --- 96 decode steps ---
  for (int t = 0; t < TT; ++t) {
    // gates = [inp|h] @ W_gates + bias  (TDM-staged B panel, 18KB LDS)
    k_gemm_wmma_tdm<<<dim3(1024 / 16), 256, KGATES * 16 * 2, stream>>>(
        inpcat, KGATES, wg, 1024, KGATES, 1024, biasg, gates, nullptr, 1024);
    k_lstm_cell<<<gblk((long)BB * HID), 256, 0, stream>>>(gates, cbuf, inpcat);
    // logits = h @ w_lin^T + b_lin  (TDM-staged B panel, 8KB LDS)
    k_gemm_wmma_tdm<<<dim3((VV + 15) / 16), 256, HID * 16 * 2, stream>>>(
        inpcat + 300, KGATES, wlt, VV, HID, VV, blin, logit, nullptr, VV);
    k_gumbel_softmax<<<BB, 256, 0, stream>>>(logit, gum, t, out, gbf);
    // nxt = gumbled @ vocab_embedding -> inpcat[:,0:300]
    k_gemm_wmma<<<dim3((EMB + 15) / 16, BB / 16), 32, 0, stream>>>(
        gbf, KFEED, vbf, EMB, KFEED, EMB, nullptr, nullptr, inpcat, KGATES);
  }
}
